// MultiHeadAttention_19404662243895
// MI455X (gfx1250) — matrix operand order repaired and checked by reading, not compiled, not run on hardware
//
#include <hip/hip_runtime.h>

// ---------------------------------------------------------------------------
// MI455X (gfx1250) multi-head attention forward.
// B=4, S=1024, D=1024, H=16, DH=64.  wave32, WMMA bf16 16x16x32, fp32 accum.
// d_out = [out (4M fp32)] ++ [qk masked logits (64M fp32)]
// ---------------------------------------------------------------------------

typedef __attribute__((ext_vector_type(16))) __bf16 v16bf;
typedef __attribute__((ext_vector_type(8)))  float  v8f;

union Frag {            // 32 bytes: one WMMA A/B operand (16 bf16 per lane)
    uint4  u[2];
    v16bf  v;
};

#define BDIM 4
#define SDIM 1024
#define DDIM 1024
#define HDIM 16
#define DH   64
#define OUT_OFF (4u * 1024u * 1024u)          // floats before qk region
#define NEG_MAX (-3.402823466e38f)            // jnp.finfo(f32).min

__device__ __forceinline__ v8f wmma_bf16(const Frag& a, const Frag& b, v8f c) {
    return __builtin_amdgcn_wmma_f32_16x16x32_bf16(
        false, a.v, false, b.v, (short)0, c, false, false);
}

// ---------------------------------------------------------------------------
// fp32 -> bf16 elementwise convert (grid-stride)
// ---------------------------------------------------------------------------
__global__ void f32_to_bf16_kernel(const float* __restrict__ src,
                                   __bf16* __restrict__ dst, int n) {
    int i = blockIdx.x * blockDim.x + threadIdx.x;
    int stride = gridDim.x * blockDim.x;
    for (; i < n; i += stride) dst[i] = (__bf16)src[i];
}

// ---------------------------------------------------------------------------
// C[m,n] = sum_k A[m,k] * W[n,k]   (M=4096, N=1024, K=1024, bf16 in, f32 acc)
// mode 0: out_bf[m*1024+n] = bf16((acc + bias) * scale)          (q / k)
// mode 2: vt layout out_bf[((b*16+h)*64+d)*1024+s] = bf16(acc+bias)  (v)
// mode 3: out_f[m*1024+n]  = acc + bias                          (o)
// Block: 256 threads = 8 waves, WG tile 64x128, wave tile 32x32.
// ---------------------------------------------------------------------------
__device__ __forceinline__ void store_proj_tile(v8f c, int mr, int nc, int lm, int lh,
                                                const float* bias, float scale,
                                                __bf16* out_bf, float* out_f, int mode) {
    int n = nc + lm;
    float bv = bias ? bias[n] : 0.0f;
#pragma unroll
    for (int r = 0; r < 8; ++r) {
        int m = mr + lh * 8 + r;
        float v = (c[r] + bv) * scale;
        if (mode == 0) {
            out_bf[(size_t)m * DDIM + n] = (__bf16)v;
        } else if (mode == 2) {
            // v transposed per head: vt[b][h][d][s]
            int b = m >> 10, s = m & 1023, h = n >> 6, d = n & 63;
            out_bf[(((size_t)(b * HDIM + h) * DH + d) * SDIM) + s] = (__bf16)v;
        } else {
            out_f[(size_t)m * DDIM + n] = v;
        }
    }
}

__global__ void gemm_xw_kernel(const __bf16* __restrict__ A,
                               const __bf16* __restrict__ W,
                               const float* __restrict__ bias, float scale,
                               __bf16* __restrict__ out_bf,
                               float* __restrict__ out_f, int mode) {
    int wave = threadIdx.x >> 5, lane = threadIdx.x & 31;
    int lm = lane & 15, lh = lane >> 4;
    int arow = blockIdx.y * 64 + (wave >> 2) * 32;
    int bcol = blockIdx.x * 128 + (wave & 3) * 32;

    v8f c00 = {}, c01 = {}, c10 = {}, c11 = {};
    const __bf16* a0 = A + (size_t)(arow + lm)      * DDIM + lh * 8;
    const __bf16* a1 = A + (size_t)(arow + 16 + lm) * DDIM + lh * 8;
    const __bf16* b0 = W + (size_t)(bcol + lm)      * DDIM + lh * 8;
    const __bf16* b1 = W + (size_t)(bcol + 16 + lm) * DDIM + lh * 8;

    for (int k0 = 0; k0 < DDIM; k0 += 32) {
        Frag fa0, fa1, fb0, fb1;
        fa0.u[0] = *(const uint4*)(a0 + k0); fa0.u[1] = *(const uint4*)(a0 + k0 + 16);
        fa1.u[0] = *(const uint4*)(a1 + k0); fa1.u[1] = *(const uint4*)(a1 + k0 + 16);
        fb0.u[0] = *(const uint4*)(b0 + k0); fb0.u[1] = *(const uint4*)(b0 + k0 + 16);
        fb1.u[0] = *(const uint4*)(b1 + k0); fb1.u[1] = *(const uint4*)(b1 + k0 + 16);
        c00 = wmma_bf16(fa0, fb0, c00);
        c01 = wmma_bf16(fa0, fb1, c01);
        c10 = wmma_bf16(fa1, fb0, c10);
        c11 = wmma_bf16(fa1, fb1, c11);
    }
    store_proj_tile(c00, arow,      bcol,      lm, lh, bias, scale, out_bf, out_f, mode);
    store_proj_tile(c01, arow,      bcol + 16, lm, lh, bias, scale, out_bf, out_f, mode);
    store_proj_tile(c10, arow + 16, bcol,      lm, lh, bias, scale, out_bf, out_f, mode);
    store_proj_tile(c11, arow + 16, bcol + 16, lm, lh, bias, scale, out_bf, out_f, mode);
}

// ---------------------------------------------------------------------------
// qk[b,h,mq,nk] = sum_d q[b,mq,h*64+d] * k[b,nk,h*64+d], masked, fp32 -> d_out
// grid (8, 16, B*H), block 256. WG tile 64x128, wave tile 32x32, K=64.
// ---------------------------------------------------------------------------
__global__ void qk_kernel(const __bf16* __restrict__ qb,
                          const __bf16* __restrict__ kb,
                          const int* __restrict__ mask,
                          float* __restrict__ qk_out) {
    int wave = threadIdx.x >> 5, lane = threadIdx.x & 31;
    int lm = lane & 15, lh = lane >> 4;
    int bh = blockIdx.z;
    int b = bh >> 4, head = bh & 15;
    int arow = blockIdx.y * 64 + (wave >> 2) * 32;
    int bcol = blockIdx.x * 128 + (wave & 3) * 32;

    const __bf16* Ab = qb + (size_t)b * SDIM * DDIM + (size_t)head * DH;
    const __bf16* Bb = kb + (size_t)b * SDIM * DDIM + (size_t)head * DH;

    v8f c00 = {}, c01 = {}, c10 = {}, c11 = {};
    const __bf16* a0 = Ab + (size_t)(arow + lm)      * DDIM + lh * 8;
    const __bf16* a1 = Ab + (size_t)(arow + 16 + lm) * DDIM + lh * 8;
    const __bf16* b0 = Bb + (size_t)(bcol + lm)      * DDIM + lh * 8;
    const __bf16* b1 = Bb + (size_t)(bcol + 16 + lm) * DDIM + lh * 8;

#pragma unroll
    for (int k0 = 0; k0 < DH; k0 += 32) {
        Frag fa0, fa1, fb0, fb1;
        fa0.u[0] = *(const uint4*)(a0 + k0); fa0.u[1] = *(const uint4*)(a0 + k0 + 16);
        fa1.u[0] = *(const uint4*)(a1 + k0); fa1.u[1] = *(const uint4*)(a1 + k0 + 16);
        fb0.u[0] = *(const uint4*)(b0 + k0); fb0.u[1] = *(const uint4*)(b0 + k0 + 16);
        fb1.u[0] = *(const uint4*)(b1 + k0); fb1.u[1] = *(const uint4*)(b1 + k0 + 16);
        c00 = wmma_bf16(fa0, fb0, c00);
        c01 = wmma_bf16(fa0, fb1, c01);
        c10 = wmma_bf16(fa1, fb0, c10);
        c11 = wmma_bf16(fa1, fb1, c11);
    }

    float* obase = qk_out + ((size_t)bh * SDIM) * SDIM;
    const int* mrow = mask + b * SDIM;
    v8f ct[4] = {c00, c01, c10, c11};
#pragma unroll
    for (int t = 0; t < 4; ++t) {
        int mr = arow + (t >> 1) * 16;
        int n  = bcol + (t & 1) * 16 + lm;
        bool pad = (mrow[n] == 0);
#pragma unroll
        for (int r = 0; r < 8; ++r) {
            int m = mr + lh * 8 + r;
            obase[(size_t)m * SDIM + n] = pad ? NEG_MAX : ct[t][r];
        }
    }
}

// ---------------------------------------------------------------------------
// Fused softmax + PV.  One WG = 32 query rows of one (b, head).
// Phase 1: softmax over qk rows (read back from d_out), probs -> bf16 LDS.
// Phase 2: wv[32,64] = w[32,1024] @ vh[1024,64] via WMMA, B-frags from vt.
// grid (S/32, H, B), block 256.
// ---------------------------------------------------------------------------
__global__ void softmax_pv_kernel(const float* __restrict__ qk,
                                  const int* __restrict__ mask,
                                  const __bf16* __restrict__ vt,
                                  __bf16* __restrict__ wvb) {
    __shared__ __align__(16) __bf16 wlds[32 * SDIM];   // 64 KB probabilities
    __shared__ float red[32][8];
    __shared__ float rowstat[32];

    int qblk = blockIdx.x, head = blockIdx.y, b = blockIdx.z;
    int tid = threadIdx.x;
    int row = tid >> 3, sub = tid & 7;                 // 8 threads per row

    const float* rp = qk + (((size_t)(b * HDIM + head) * SDIM) + qblk * 32 + row) * SDIM
                         + sub * 128;
    const int* mrow = mask + b * SDIM + sub * 128;

    // pass 1: row max
    float mx = NEG_MAX;
    for (int i = 0; i < 128; ++i) mx = fmaxf(mx, rp[i]);
    red[row][sub] = mx;
    __syncthreads();
    if (sub == 0) {
        float m2 = red[row][0];
        for (int i = 1; i < 8; ++i) m2 = fmaxf(m2, red[row][i]);
        rowstat[row] = m2;
    }
    __syncthreads();
    float rmax = rowstat[row];
    __syncthreads();

    // pass 2: exp, zero pads, partial sums, stash bf16 probs
    __bf16* wp = wlds + row * SDIM + sub * 128;
    float s = 0.0f;
    for (int i = 0; i < 128; ++i) {
        float e = (mrow[i] != 0) ? __expf(rp[i] - rmax) : 0.0f;
        s += e;
        wp[i] = (__bf16)e;
    }
    red[row][sub] = s;
    __syncthreads();
    if (sub == 0) {
        float s2 = 0.0f;
        for (int i = 0; i < 8; ++i) s2 += red[row][i];
        rowstat[row] = (s2 > 0.0f) ? (1.0f / s2) : 0.0f;
    }
    __syncthreads();
    float inv = rowstat[row];
    for (int i = 0; i < 128; ++i) wp[i] = (__bf16)((float)wp[i] * inv);
    __syncthreads();

    // ---- PV GEMM: each wave does one 16x16 tile of the 32x64 result ----
    int wave = tid >> 5, lane = tid & 31;
    int lm = lane & 15, lh = lane >> 4;
    int tm = (wave & 1) * 16;          // 0 / 16 (query rows)
    int tn = (wave >> 1) * 16;         // 0 / 16 / 32 / 48 (head dims)

    const __bf16* vbase = vt + ((size_t)(b * HDIM + head) * DH + tn + lm) * SDIM + lh * 8;
    const __bf16* abase = wlds + (tm + lm) * SDIM + lh * 8;

    v8f acc = {};
    for (int k0 = 0; k0 < SDIM; k0 += 32) {
        Frag fa, fb;
        fa.u[0] = *(const uint4*)(abase + k0);
        fa.u[1] = *(const uint4*)(abase + k0 + 16);
        fb.u[0] = *(const uint4*)(vbase + k0);
        fb.u[1] = *(const uint4*)(vbase + k0 + 16);
        acc = wmma_bf16(fa, fb, acc);
    }

    int m0 = qblk * 32 + tm + lh * 8;
    int col = head * DH + tn + lm;
    __bf16* op = wvb + ((size_t)(b * SDIM + m0)) * DDIM + col;
#pragma unroll
    for (int r = 0; r < 8; ++r) op[(size_t)r * DDIM] = (__bf16)acc[r];
}

// ---------------------------------------------------------------------------
extern "C" void kernel_launch(void* const* d_in, const int* in_sizes, int n_in,
                              void* d_out, int out_size, void* d_ws, size_t ws_size,
                              hipStream_t stream) {
    const float* x    = (const float*)d_in[0];
    const int*   mask = (const int*)  d_in[1];
    const float* Wq   = (const float*)d_in[2];
    const float* bq   = (const float*)d_in[3];
    const float* Wk   = (const float*)d_in[4];
    const float* Wv   = (const float*)d_in[5];
    const float* bv   = (const float*)d_in[6];
    const float* Wo   = (const float*)d_in[7];
    const float* bo   = (const float*)d_in[8];

    float* out_f  = (float*)d_out;                 // [4,1024,1024]
    float* qk_out = (float*)d_out + OUT_OFF;       // [4,16,1024,1024]

    // workspace layout (bytes)
    char* w = (char*)d_ws;
    __bf16* xb  = (__bf16*)(w);                           //  8 MB  x bf16
    __bf16* Wqb = (__bf16*)(w + ( 8u << 20));             //  2 MB
    __bf16* Wkb = (__bf16*)(w + (10u << 20));             //  2 MB
    __bf16* Wvb = (__bf16*)(w + (12u << 20));             //  2 MB
    __bf16* Wob = (__bf16*)(w + (14u << 20));             //  2 MB
    __bf16* qb  = (__bf16*)(w + (16u << 20));             //  8 MB  scaled q
    __bf16* kb  = (__bf16*)(w + (24u << 20));             //  8 MB  scaled k
    __bf16* vt  = (__bf16*)(w + (32u << 20));             //  8 MB  v^T per head
    __bf16* wvb = (__bf16*)(w + (40u << 20));             //  8 MB  attn output

    const int NX = BDIM * SDIM * DDIM;    // 4M
    const int NW = DDIM * DDIM;           // 1M
    f32_to_bf16_kernel<<<1024, 256, 0, stream>>>(x,  xb,  NX);
    f32_to_bf16_kernel<<<1024, 256, 0, stream>>>(Wq, Wqb, NW);
    f32_to_bf16_kernel<<<1024, 256, 0, stream>>>(Wk, Wkb, NW);
    f32_to_bf16_kernel<<<1024, 256, 0, stream>>>(Wv, Wvb, NW);
    f32_to_bf16_kernel<<<1024, 256, 0, stream>>>(Wo, Wob, NW);

    const float scale = 0.3535533905932738f;      // DH^-0.25, folded into q and k
    dim3 ggrid(8, 64, 1);                          // N/128 x M/64
    // q = x@Wq^T + bq, scaled
    gemm_xw_kernel<<<ggrid, 256, 0, stream>>>(xb, Wqb, bq, scale, qb, nullptr, 0);
    // k = x@Wk^T, scaled (no bias)
    gemm_xw_kernel<<<ggrid, 256, 0, stream>>>(xb, Wkb, nullptr, scale, kb, nullptr, 0);
    // v = x@Wv^T + bv, stored transposed per head
    gemm_xw_kernel<<<ggrid, 256, 0, stream>>>(xb, Wvb, bv, 1.0f, vt, nullptr, 2);

    // masked logits -> d_out qk region
    qk_kernel<<<dim3(8, 16, BDIM * HDIM), 256, 0, stream>>>(qb, kb, mask, qk_out);

    // fused softmax + PV -> wvb
    softmax_pv_kernel<<<dim3(SDIM / 32, HDIM, BDIM), 256, 0, stream>>>(qk_out, mask, vt, wvb);

    // out = wv@Wo^T + bo (fp32)
    gemm_xw_kernel<<<ggrid, 256, 0, stream>>>(wvb, Wob, bo, 1.0f, nullptr, out_f, 3);
}
